// LanguageModel_87230785781887
// MI455X (gfx1250) — matrix operand order repaired and checked by reading, not compiled, not run on hardware
//
#include <hip/hip_runtime.h>

typedef __attribute__((ext_vector_type(16))) _Float16 v16h;
typedef __attribute__((ext_vector_type(8)))  _Float16 v8h;
typedef __attribute__((ext_vector_type(8)))  float    v8f;

static constexpr int Bn = 64, Tn = 256, Cn = 256, Hn = 8, Ln = 6, Vn = 96;
static constexpr int HSn = Cn / Hn;   // 32
static constexpr int BT  = Bn * Tn;   // 16384

union V16 { v16h v; v8h h[2]; _Float16 e[16]; };

__device__ inline v8f wmma16(v16h a, v16h b, v8f c) {
  // v_wmma_f32_16x16x32_f16: D = A(16x32) * B(32x16) + C(16x16 f32)
  return __builtin_amdgcn_wmma_f32_16x16x32_f16(false, a, false, b, (short)0, c,
                                                false, false);
}

// ---------------------------------------------------------------------------
// f32 -> f16 conversion (weights)
// ---------------------------------------------------------------------------
__global__ void cvt_f32_f16_kernel(const float* __restrict__ in,
                                   _Float16* __restrict__ out, int n) {
  int i = blockIdx.x * blockDim.x + threadIdx.x;
  if (i < n) out[i] = (_Float16)in[i];
}

// ---------------------------------------------------------------------------
// Embedding: x = tok_emb[idx] + pos_emb, write f32 residual + f16 operand
// ---------------------------------------------------------------------------
__global__ void embed_kernel(const int* __restrict__ idx,
                             const float* __restrict__ tok,
                             const float* __restrict__ pos,
                             float* __restrict__ x, _Float16* __restrict__ xh) {
  int i = blockIdx.x * blockDim.x + threadIdx.x;  // over BT*Cn
  int c = i & (Cn - 1);
  int bt = i >> 8;              // Cn == 256
  int t = bt & (Tn - 1);        // Tn == 256
  float v = tok[(size_t)idx[bt] * Cn + c] + pos[(size_t)t * Cn + c];
  x[i] = v;
  xh[i] = (_Float16)v;
}

// ---------------------------------------------------------------------------
// WMMA GEMM, register-blocked: each wave computes a 32x32 output tile as
// 2x2 16x16 WMMA fragments (4 independent accumulators). Per 32-deep K chunk:
// 4 A loads + 4 B loads feed 4 WMMAs (32 B of loads per WMMA).
// out[M,N] = act(A[M,K=256] * W[K,N] + bias[N])
// ---------------------------------------------------------------------------
template <bool RELU, bool OUTF16>
__global__ __launch_bounds__(128) void gemm_wmma_kernel(
    const _Float16* __restrict__ A, const _Float16* __restrict__ W,
    const float* __restrict__ bias, void* __restrict__ out, int N) {
  const int lane = threadIdx.x & 31;
  const int wave = blockIdx.x * 4 + (threadIdx.x >> 5);
  const int NB = N >> 5;                 // 32-wide N blocks
  const int mb = wave / NB;
  const int nb = wave - mb * NB;
  const int m0 = mb * 32;
  const int n0 = nb * 32;
  const int mloc = lane & 15;
  const int hh = lane >> 4;

  const _Float16* arow0 = A + (size_t)(m0 + mloc) * Cn;        // rows m0..+15
  const _Float16* arow1 = arow0 + (size_t)16 * Cn;             // rows m0+16..+31
  const _Float16* wb = W + n0 + mloc;                          // column = n0 + lane%16

  v8f acc00 = {}, acc01 = {}, acc10 = {}, acc11 = {};
#pragma unroll
  for (int k0 = 0; k0 < Cn; k0 += 32) {
    V16 a0, a1, b0, b1;
    a0.h[0] = *(const v8h*)(arow0 + k0 + 8 * hh);       // K = 8*hh .. +7
    a0.h[1] = *(const v8h*)(arow0 + k0 + 16 + 8 * hh);  // K = 16+8*hh .. +7
    a1.h[0] = *(const v8h*)(arow1 + k0 + 8 * hh);
    a1.h[1] = *(const v8h*)(arow1 + k0 + 16 + 8 * hh);
    const _Float16* wk = wb + (size_t)k0 * N;
#pragma unroll
    for (int s = 0; s < 16; ++s) {
      const int kk = (s < 8) ? (8 * hh + s) : (16 + 8 * hh + s - 8);
      b0.e[s] = wk[(size_t)kk * N];
      b1.e[s] = wk[(size_t)kk * N + 16];
    }
    acc00 = wmma16(a0.v, b0.v, acc00);
    acc01 = wmma16(a0.v, b1.v, acc01);
    acc10 = wmma16(a1.v, b0.v, acc10);
    acc11 = wmma16(a1.v, b1.v, acc11);
  }

  const int col0 = n0 + mloc;
  const int col1 = col0 + 16;
  const float bv0 = bias ? bias[col0] : 0.0f;
  const float bv1 = bias ? bias[col1] : 0.0f;
#pragma unroll
  for (int r = 0; r < 8; ++r) {
    const int row0 = m0 + r + hh * 8;   // C/D layout: M = r + 8*(lane>=16)
    const int row1 = row0 + 16;
    float v00 = acc00[r] + bv0, v01 = acc01[r] + bv1;
    float v10 = acc10[r] + bv0, v11 = acc11[r] + bv1;
    if (RELU) {
      v00 = fmaxf(v00, 0.0f); v01 = fmaxf(v01, 0.0f);
      v10 = fmaxf(v10, 0.0f); v11 = fmaxf(v11, 0.0f);
    }
    if (OUTF16) {
      _Float16* o = (_Float16*)out;
      o[(size_t)row0 * N + col0] = (_Float16)v00;
      o[(size_t)row0 * N + col1] = (_Float16)v01;
      o[(size_t)row1 * N + col0] = (_Float16)v10;
      o[(size_t)row1 * N + col1] = (_Float16)v11;
    } else {
      float* o = (float*)out;
      o[(size_t)row0 * N + col0] = v00;
      o[(size_t)row0 * N + col1] = v01;
      o[(size_t)row1 * N + col0] = v10;
      o[(size_t)row1 * N + col1] = v11;
    }
  }
}

// ---------------------------------------------------------------------------
// Fused causal attention for one (b, h) per block. 4 waves; each wave owns
// four 16-query tiles. Scores via WMMA (Q x K^T from LDS), masked softmax on
// f16 score tiles in LDS, then PV via 2 WMMAs per 32-key chunk.
// ---------------------------------------------------------------------------
__global__ __launch_bounds__(128) void attn_wmma_kernel(
    const _Float16* __restrict__ qh, const _Float16* __restrict__ kh,
    const _Float16* __restrict__ vh, _Float16* __restrict__ atth) {
  const int b = blockIdx.x >> 3;   // Hn == 8
  const int h = blockIdx.x & 7;
  const int lane = threadIdx.x & 31;
  const int wv = threadIdx.x >> 5;
  const float scale = 0.0625f;     // Cn^-0.5 (reference scales by n_embed)

  __shared__ __align__(16) _Float16 Kt[HSn][Tn];       // K transposed: 16 KB
  __shared__ __align__(16) _Float16 P[4][16][Tn];      // per-wave tiles: 32 KB
  __shared__ float rowmax[4][16];

  // Stage K^T (d-major) into LDS: vectorized global read, scattered LDS write
  const size_t basek = ((size_t)b * Tn) * Cn + h * HSn;
  for (int i = threadIdx.x; i < Tn * 4; i += blockDim.x) {
    int t = i >> 2, d0 = (i & 3) * 8;
    v8h kv = *(const v8h*)(kh + basek + (size_t)t * Cn + d0);
#pragma unroll
    for (int j = 0; j < 8; ++j) Kt[d0 + j][t] = kv[j];
  }
  __syncthreads();

  const int m = lane & 15;
  const int hh = lane >> 4;

  for (int qt = wv; qt < 16; qt += 4) {
    // --- Q tile in WMMA A layout (contract dim = head dim, 32) ---
    V16 aq;
    const _Float16* qrow = qh + ((size_t)b * Tn + qt * 16 + m) * Cn + h * HSn;
    aq.h[0] = *(const v8h*)(qrow + 8 * hh);
    aq.h[1] = *(const v8h*)(qrow + 16 + 8 * hh);

    float lmax[8];
#pragma unroll
    for (int r = 0; r < 8; ++r) lmax[r] = -3.0e38f;

    // --- scores: S = scale * Q K^T, causal mask, stash f16 tiles in LDS ---
    for (int kt = 0; kt <= qt; ++kt) {
      V16 bk;
#pragma unroll
      for (int s = 0; s < 16; ++s) {
        const int kk = (s < 8) ? (8 * hh + s) : (16 + 8 * hh + s - 8);
        bk.e[s] = Kt[kk][kt * 16 + m];
      }
      v8f s = {};
      s = wmma16(aq.v, bk.v, s);
#pragma unroll
      for (int r = 0; r < 8; ++r) {
        int qi = qt * 16 + r + hh * 8;
        int ki = kt * 16 + m;
        float sv = (ki <= qi) ? s[r] * scale : -3.0e38f;
        P[wv][r + hh * 8][kt * 16 + m] = (_Float16)sv;  // overflows to -inf
        lmax[r] = fmaxf(lmax[r], sv);
      }
    }
    // tail (never-attended keys) -> -inf so exp() gives exactly 0
    for (int t = qt * 16 + 16 + lane; t < Tn; t += 32) {
#pragma unroll
      for (int r2 = 0; r2 < 16; ++r2) P[wv][r2][t] = (_Float16)(-3.0e38f);
    }

    // row max across the 16 lanes sharing each row
#pragma unroll
    for (int r = 0; r < 8; ++r) {
      float v = lmax[r];
      v = fmaxf(v, __shfl_xor(v, 1, 32));
      v = fmaxf(v, __shfl_xor(v, 2, 32));
      v = fmaxf(v, __shfl_xor(v, 4, 32));
      v = fmaxf(v, __shfl_xor(v, 8, 32));
      if ((lane & 15) == 0) rowmax[wv][r + hh * 8] = v;
    }

    // --- softmax over each of the 16 rows (full wave per row) ---
    for (int r2 = 0; r2 < 16; ++r2) {
      float mx = rowmax[wv][r2];
      float accv = 0.0f;
      for (int t = lane; t < Tn; t += 32) {
        float e = __expf((float)P[wv][r2][t] - mx);
        P[wv][r2][t] = (_Float16)e;    // e in (0, 1]
        accv += e;
      }
#pragma unroll
      for (int off = 1; off < 32; off <<= 1) accv += __shfl_xor(accv, off, 32);
      float rinv = 1.0f / accv;
      for (int t = lane; t < Tn; t += 32)
        P[wv][r2][t] = (_Float16)((float)P[wv][r2][t] * rinv);
    }

    // --- O = P V : two 16x16 halves of the 32-wide head dim per key chunk ---
    v8f o0 = {}, o1 = {};
    const int nc = (qt * 16 + 16 + 31) >> 5;   // only chunks that can attend
    for (int c = 0; c < nc; ++c) {
      V16 ap;
      ap.h[0] = *(const v8h*)&P[wv][m][c * 32 + 8 * hh];
      ap.h[1] = *(const v8h*)&P[wv][m][c * 32 + 16 + 8 * hh];
      V16 bv0, bv1;
#pragma unroll
      for (int s = 0; s < 16; ++s) {
        const int kk = (s < 8) ? (8 * hh + s) : (16 + 8 * hh + s - 8);
        const _Float16* vrow = vh + ((size_t)b * Tn + c * 32 + kk) * Cn + h * HSn + m;
        bv0.e[s] = vrow[0];
        bv1.e[s] = vrow[16];
      }
      o0 = wmma16(ap.v, bv0.v, o0);
      o1 = wmma16(ap.v, bv1.v, o1);
    }
#pragma unroll
    for (int r = 0; r < 8; ++r) {
      int tq = qt * 16 + r + hh * 8;
      _Float16* orow = atth + ((size_t)b * Tn + tq) * Cn + h * HSn + m;
      orow[0]  = (_Float16)o0[r];
      orow[16] = (_Float16)o1[r];
    }
  }
}

// ---------------------------------------------------------------------------
// LayerNorm: y = LN(xin (+ res)) * w + b -> f32 (optional) + f16
// ---------------------------------------------------------------------------
__global__ __launch_bounds__(256) void ln_kernel(
    const float* __restrict__ xin, const float* __restrict__ res,
    const float* __restrict__ w, const float* __restrict__ bias,
    float* __restrict__ xout, _Float16* __restrict__ xh) {
  const int row = blockIdx.x;
  const int t = threadIdx.x;
  const size_t i = (size_t)row * Cn + t;
  float v = xin[i] + (res ? res[i] : 0.0f);
  float s = v, s2 = v * v;
#pragma unroll
  for (int off = 1; off < 32; off <<= 1) {
    s  += __shfl_xor(s, off, 32);
    s2 += __shfl_xor(s2, off, 32);
  }
  __shared__ float sred[8], s2red[8];
  if ((t & 31) == 0) { sred[t >> 5] = s; s2red[t >> 5] = s2; }
  __syncthreads();
  float tot = 0.0f, tot2 = 0.0f;
#pragma unroll
  for (int j = 0; j < 8; ++j) { tot += sred[j]; tot2 += s2red[j]; }
  const float mean = tot * (1.0f / Cn);
  const float var = tot2 * (1.0f / Cn) - mean * mean;
  const float inv = rsqrtf(var + 1e-5f);
  const float y = (v - mean) * inv * w[t] + bias[t];
  if (xout) xout[i] = y;
  xh[i] = (_Float16)y;
}

// ---------------------------------------------------------------------------
// Loss: per-row NLL then deterministic reduction
// ---------------------------------------------------------------------------
__global__ __launch_bounds__(256) void loss_row_kernel(
    const float* __restrict__ logits, const int* __restrict__ target,
    float* __restrict__ rownll) {
  const int row = blockIdx.x * 8 + (threadIdx.x >> 5);
  const int lane = threadIdx.x & 31;
  const float* lr = logits + (size_t)row * Vn;
  float mx = -3.0e38f;
  for (int j = lane; j < Vn; j += 32) mx = fmaxf(mx, lr[j]);
#pragma unroll
  for (int off = 1; off < 32; off <<= 1) mx = fmaxf(mx, __shfl_xor(mx, off, 32));
  float s = 0.0f;
  for (int j = lane; j < Vn; j += 32) s += __expf(lr[j] - mx);
#pragma unroll
  for (int off = 1; off < 32; off <<= 1) s += __shfl_xor(s, off, 32);
  if (lane == 0)
    rownll[row] = -(lr[target[row]] - mx - __logf(s));
}

__global__ __launch_bounds__(256) void loss_reduce_kernel(
    const float* __restrict__ rownll, float* __restrict__ loss_out) {
  __shared__ float red[256];
  float acc = 0.0f;
  for (int i = threadIdx.x; i < BT; i += 256) acc += rownll[i];
  red[threadIdx.x] = acc;
  __syncthreads();
  for (int sdist = 128; sdist > 0; sdist >>= 1) {
    if (threadIdx.x < sdist) red[threadIdx.x] += red[threadIdx.x + sdist];
    __syncthreads();
  }
  if (threadIdx.x == 0) *loss_out = red[0] * (1.0f / BT);
}

// ---------------------------------------------------------------------------
extern "C" void kernel_launch(void* const* d_in, const int* in_sizes, int n_in,
                              void* d_out, int out_size, void* d_ws,
                              size_t ws_size, hipStream_t stream) {
  (void)in_sizes; (void)n_in; (void)out_size; (void)ws_size;
  const int*   idx    = (const int*)d_in[0];
  const int*   target = (const int*)d_in[1];
  const float* tok    = (const float*)d_in[2];
  const float* pos    = (const float*)d_in[3];
  const float* Wq     = (const float*)d_in[4];
  const float* Wk     = (const float*)d_in[5];
  const float* Wv     = (const float*)d_in[6];
  const float* Wo     = (const float*)d_in[7];
  const float* bo     = (const float*)d_in[8];
  const float* W1     = (const float*)d_in[9];
  const float* b1     = (const float*)d_in[10];
  const float* W2     = (const float*)d_in[11];
  const float* b2     = (const float*)d_in[12];
  const float* ln1w   = (const float*)d_in[13];
  const float* ln1b   = (const float*)d_in[14];
  const float* ln2w   = (const float*)d_in[15];
  const float* ln2b   = (const float*)d_in[16];
  const float* lnfw   = (const float*)d_in[17];
  const float* lnfb   = (const float*)d_in[18];
  const float* Wlm    = (const float*)d_in[19];
  const float* blm    = (const float*)d_in[20];

  char* wsp = (char*)d_ws;
  auto take = [&](size_t bytes) {
    char* p = wsp;
    wsp += (bytes + 255) & ~(size_t)255;
    return p;
  };
  float*    x    = (float*)take((size_t)BT * Cn * 4);
  float*    tmp  = (float*)take((size_t)BT * Cn * 4);
  _Float16* xh   = (_Float16*)take((size_t)BT * Cn * 2);
  _Float16* qh   = (_Float16*)take((size_t)BT * Cn * 2);
  _Float16* kh   = (_Float16*)take((size_t)BT * Cn * 2);
  _Float16* vh   = (_Float16*)take((size_t)BT * Cn * 2);
  _Float16* ah   = (_Float16*)take((size_t)BT * Cn * 2);
  _Float16* h1h  = (_Float16*)take((size_t)BT * Cn * 2);
  _Float16* wqh  = (_Float16*)take((size_t)Ln * Cn * Cn * 2);
  _Float16* wkh  = (_Float16*)take((size_t)Ln * Cn * Cn * 2);
  _Float16* wvh  = (_Float16*)take((size_t)Ln * Cn * Cn * 2);
  _Float16* woh  = (_Float16*)take((size_t)Ln * Cn * Cn * 2);
  _Float16* w1h  = (_Float16*)take((size_t)Ln * Cn * Cn * 2);
  _Float16* w2h  = (_Float16*)take((size_t)Ln * Cn * Cn * 2);
  _Float16* wlmh = (_Float16*)take((size_t)Cn * Vn * 2);
  float*    rnll = (float*)take((size_t)BT * 4);

  // weight conversion (deterministic, graph-safe)
  const int wn = Ln * Cn * Cn;
  const int wblk = (wn + 255) / 256;
  cvt_f32_f16_kernel<<<wblk, 256, 0, stream>>>(Wq, wqh, wn);
  cvt_f32_f16_kernel<<<wblk, 256, 0, stream>>>(Wk, wkh, wn);
  cvt_f32_f16_kernel<<<wblk, 256, 0, stream>>>(Wv, wvh, wn);
  cvt_f32_f16_kernel<<<wblk, 256, 0, stream>>>(Wo, woh, wn);
  cvt_f32_f16_kernel<<<wblk, 256, 0, stream>>>(W1, w1h, wn);
  cvt_f32_f16_kernel<<<wblk, 256, 0, stream>>>(W2, w2h, wn);
  cvt_f32_f16_kernel<<<(Cn * Vn + 255) / 256, 256, 0, stream>>>(Wlm, wlmh, Cn * Vn);

  embed_kernel<<<BT * Cn / 256, 256, 0, stream>>>(idx, tok, pos, x, xh);

  // 32x32-tile GEMM grids: waves = (M/32)*(N/32), 4 waves per block
  const int gemmBlocksC = (BT / 32) * (Cn / 32) / 4;   // 1024 blocks
  for (int l = 0; l < Ln; ++l) {
    const size_t wl = (size_t)l * Cn * Cn;
    gemm_wmma_kernel<false, true><<<gemmBlocksC, 128, 0, stream>>>(
        xh, wqh + wl, nullptr, qh, Cn);
    gemm_wmma_kernel<false, true><<<gemmBlocksC, 128, 0, stream>>>(
        xh, wkh + wl, nullptr, kh, Cn);
    gemm_wmma_kernel<false, true><<<gemmBlocksC, 128, 0, stream>>>(
        xh, wvh + wl, nullptr, vh, Cn);
    attn_wmma_kernel<<<Bn * Hn, 128, 0, stream>>>(qh, kh, vh, ah);
    gemm_wmma_kernel<false, false><<<gemmBlocksC, 128, 0, stream>>>(
        ah, woh + wl, bo + (size_t)l * Cn, tmp, Cn);
    ln_kernel<<<BT, 256, 0, stream>>>(x, tmp, ln1w + (size_t)l * Cn,
                                      ln1b + (size_t)l * Cn, x, xh);
    gemm_wmma_kernel<true, true><<<gemmBlocksC, 128, 0, stream>>>(
        xh, w1h + wl, b1 + (size_t)l * Cn, h1h, Cn);
    gemm_wmma_kernel<false, false><<<gemmBlocksC, 128, 0, stream>>>(
        h1h, w2h + wl, b2 + (size_t)l * Cn, tmp, Cn);
    ln_kernel<<<BT, 256, 0, stream>>>(x, tmp, ln2w + (size_t)l * Cn,
                                      ln2b + (size_t)l * Cn, x, xh);
  }
  // final LN (no residual), f16 only
  ln_kernel<<<BT, 256, 0, stream>>>(x, nullptr, lnfw, lnfb, nullptr, xh);

  // LM head -> logits (f32, straight into d_out)
  float* logits = (float*)d_out;
  const int gemmBlocksV = (BT / 32) * (Vn / 32) / 4;   // 384 blocks
  gemm_wmma_kernel<false, false><<<gemmBlocksV, 128, 0, stream>>>(
      xh, wlmh, blm, logits, Vn);

  // loss (deterministic two-stage reduction), written after the logits
  loss_row_kernel<<<BT / 8, 256, 0, stream>>>(logits, target, rnll);
  loss_reduce_kernel<<<1, 256, 0, stream>>>(rnll, logits + (size_t)BT * Vn);
}
